// PPAttention_69853348102639
// MI455X (gfx1250) — compile-verified, matrix operand order checked by reading, not run on hardware
//
#include <hip/hip_runtime.h>

// ---------------------------------------------------------------------------
// PPAttention on MI455X (gfx1250, wave32).
// All GEMMs via v_wmma_f32_16x16x32_bf16 (bf16 in, f32 accumulate).
// N=8, H=W=128 (HW=16384), C=256, P=16.
// ---------------------------------------------------------------------------

#define HW_DIM 16384
#define C_DIM  256
#define P_DIM  16

typedef __attribute__((ext_vector_type(16))) __bf16        v16bf;
typedef __attribute__((ext_vector_type(8)))  float         v8f;
typedef __attribute__((ext_vector_type(4)))  unsigned int  u32x4;

union FragU { v16bf v; u32x4 q[2]; };

// ---- f32 -> bf16 conversion -----------------------------------------------
__device__ __forceinline__ unsigned short f2bf_sw(float f) {
  unsigned u = __float_as_uint(f);
  return (unsigned short)((u + 0x7FFFu + ((u >> 16) & 1u)) >> 16);
}

// pack two f32 -> two bf16 in one uint (v_cvt_pk_bf16_f32 when available)
__device__ __forceinline__ unsigned int pk2(float a, float b) {
#if __has_builtin(__builtin_amdgcn_cvt_pk_bf16_f32)
  typedef __attribute__((ext_vector_type(2))) __bf16 v2bf;
  v2bf p = __builtin_amdgcn_cvt_pk_bf16_f32(a, b);
  return __builtin_bit_cast(unsigned int, p);
#else
  return (unsigned)f2bf_sw(a) | ((unsigned)f2bf_sw(b) << 16);
#endif
}
__device__ __forceinline__ unsigned short f2bf(float f) {
  return (unsigned short)(pk2(f, f) & 0xFFFFu);
}

// ---- xor-lane exchange within 16-lane halves via ds_swizzle ----------------
// group-of-32 mode: lane' = ((lane & 0x1f) | 0) ^ XMASK  -> single DS op,
// no mbcnt/cndmask preamble like HIP's generic __shfl_xor.
template <int XMASK>
__device__ __forceinline__ float swz_xor(float v) {
  int r = __builtin_amdgcn_ds_swizzle(__float_as_int(v), (XMASK << 10) | 0x1F);
  return __int_as_float(r);
}
__device__ __forceinline__ float red_max16(float v) {
  v = fmaxf(v, swz_xor<1>(v));
  v = fmaxf(v, swz_xor<2>(v));
  v = fmaxf(v, swz_xor<4>(v));
  v = fmaxf(v, swz_xor<8>(v));
  return v;
}
__device__ __forceinline__ float red_sum16(float v) {
  v += swz_xor<1>(v);
  v += swz_xor<2>(v);
  v += swz_xor<4>(v);
  v += swz_xor<8>(v);
  return v;
}

// ---- WMMA fragment loads ---------------------------------------------------
// A/B fragment (16x32 / 32x16 bf16) from LDS, row-major with `pitch` elements.
// Per ISA layout: lane L -> row/col = L%16,
// k = (L/16)*8 + (elem/8)*16 + elem%8  => two contiguous 16B loads per lane.
__device__ __forceinline__ v16bf frag_ld(const unsigned short* base, int pitch,
                                         int row0, int lane, int kb) {
  int g = (lane >> 4) & 1, m = lane & 15;
  const unsigned short* p = base + (row0 + m) * pitch + kb + (g << 3);
  FragU u;
  u.q[0] = *(const u32x4*)p;
  u.q[1] = *(const u32x4*)(p + 16);
  return u.v;
}

// Fragment with only K=0..15 real data (K=16..31 zero) for the K=16 GEMM.
__device__ __forceinline__ v16bf frag_ld_half(const unsigned short* base,
                                              int pitch, int row0, int lane) {
  int g = (lane >> 4) & 1, m = lane & 15;
  const unsigned short* p = base + (row0 + m) * pitch + (g << 3);
  FragU u;
  u32x4 zq = {0u, 0u, 0u, 0u};
  u.q[0] = *(const u32x4*)p;
  u.q[1] = zq;
  return u.v;
}

__device__ __forceinline__ v8f wmma_bf16(v16bf a, v16bf b, v8f c) {
  return __builtin_amdgcn_wmma_f32_16x16x32_bf16(false, a, false, b,
                                                 (short)0, c, false, false);
}

// ---------------------------------------------------------------------------
// Kernel 0: zero the f32 keys accumulator (8*16*256 floats)
// ---------------------------------------------------------------------------
__global__ void pp_zero(float* __restrict__ p) {
  p[blockIdx.x * 256 + threadIdx.x] = 0.0f;
}

// ---------------------------------------------------------------------------
// Kernel 1: logits -> softmax(y) -> keys += y^T @ x   (atomic f32 into d_ws)
// 64 threads (2 waves), 32-pixel tile per workgroup. grid = 8*512.
// ---------------------------------------------------------------------------
__global__ __launch_bounds__(64) void pp_keys(const float* __restrict__ xg,
                                              const float* __restrict__ wg,
                                              float* __restrict__ keysF) {
  __shared__ unsigned short Xt [32 * 264];   // x tile, row = pixel   (A frags)
  __shared__ unsigned short XtT[256 * 40];   // x tile, row = channel (B frags)
  __shared__ unsigned short Wt [16 * 264];   // W_cls^T, row = p      (B frags)
  __shared__ unsigned short yT [16 * 40];    // y^T, row = p          (A frags)

  const int tid  = threadIdx.x;
  const int lane = tid & 31;
  const int wv   = tid >> 5;
  const int n    = blockIdx.x >> 9;          // 512 tiles per batch
  const int m0   = (blockIdx.x & 511) << 5;  // first pixel of tile

  // stage x tile (32 x 256 f32 -> bf16), coalesced float4 loads
  const float4* xrow = (const float4*)(xg + (size_t)(n * HW_DIM + m0) * C_DIM);
  unsigned int* XtU = (unsigned int*)Xt;
  for (int idx = tid; idx < 32 * 64; idx += 64) {
    int row = idx >> 6, c4 = idx & 63;
    float4 f = xrow[row * 64 + c4];
    unsigned u01 = pk2(f.x, f.y);
    unsigned u23 = pk2(f.z, f.w);
    int base = row * 264 + c4 * 4;
    XtU[(base >> 1) + 0] = u01;
    XtU[(base >> 1) + 1] = u23;
    int c = c4 * 4;
    XtT[(c + 0) * 40 + row] = (unsigned short)(u01 & 0xFFFFu);
    XtT[(c + 1) * 40 + row] = (unsigned short)(u01 >> 16);
    XtT[(c + 2) * 40 + row] = (unsigned short)(u23 & 0xFFFFu);
    XtT[(c + 3) * 40 + row] = (unsigned short)(u23 >> 16);
  }
  // stage W_cls (C=256 x P=16, row-major) transposed: Wt[p][c]
  const float2* wg2 = (const float2*)wg;
  for (int idx = tid; idx < (C_DIM * P_DIM) / 2; idx += 64) {
    int c = idx >> 3, p2 = idx & 7;          // two consecutive p per thread
    float2 w = wg2[idx];
    unsigned u = pk2(w.x, w.y);
    Wt[(p2 * 2 + 0) * 264 + c] = (unsigned short)(u & 0xFFFFu);
    Wt[(p2 * 2 + 1) * 264 + c] = (unsigned short)(u >> 16);
  }
  __syncthreads();

  // logits(16pix x 16p) = Xt @ Wcls, K = 256 in 8 WMMA steps
  const int r0 = wv << 4;
  const int g = lane >> 4, m = lane & 15;
  v8f acc = {0, 0, 0, 0, 0, 0, 0, 0};
#pragma unroll
  for (int kc = 0; kc < 8; ++kc) {
    v16bf a = frag_ld(Xt, 264, r0, lane, kc * 32);
    v16bf b = frag_ld(Wt, 264, 0, lane, kc * 32);
    acc = wmma_bf16(a, b, acc);
  }

  // softmax over p: D layout = one pixel-row per VGPR across each 16-lane half
#pragma unroll
  for (int r = 0; r < 8; ++r) {
    float v = acc[r];
    float mx = red_max16(v);
    float e = __expf(v - mx);
    float s = red_sum16(e);
    float y = e / s;
    int pix = r0 + r + (g << 3);               // pixel index within tile
    yT[m * 40 + pix] = f2bf(y);                // yT[p][pix]
  }
  __syncthreads();

  // keys_partial(16p x 256c) = yT(16 x 32pix) @ x(32pix x 256c)
  v16bf a_y = frag_ld(yT, 40, 0, lane, 0);
#pragma unroll
  for (int t = 0; t < 8; ++t) {
    int ct = (wv << 3) + t;                    // 16 c-tiles split over 2 waves
    v16bf b = frag_ld(XtT, 40, ct << 4, lane, 0);
    v8f z = {0, 0, 0, 0, 0, 0, 0, 0};
    v8f k = wmma_bf16(a_y, b, z);
#pragma unroll
    for (int r = 0; r < 8; ++r) {
      int p = r + (g << 3);
      int c = (ct << 4) + m;
      atomicAdd(&keysF[(size_t)((n << 4) + p) * C_DIM + c], k[r]);
    }
  }
}

// ---------------------------------------------------------------------------
// Kernel 2: keys f32 -> bf16 in (P,C) and (C,P) layouts
// ---------------------------------------------------------------------------
__global__ void pp_cvt(const float* __restrict__ keysF,
                       unsigned short* __restrict__ keysB,
                       unsigned short* __restrict__ keysT) {
  int i = blockIdx.x * 256 + threadIdx.x;     // 8*16*256 = 32768
  float v = keysF[i];
  unsigned short h = f2bf(v);
  keysB[i] = h;                               // [n][p][c]
  int n = i >> 12, rem = i & 4095, p = rem >> 8, c = rem & 255;
  keysT[(n << 12) + (c << 4) + p] = h;        // [n][c][p]
}

// ---------------------------------------------------------------------------
// Kernel 3: sim = softmax(x @ keys^T / 16); out = sim @ keys
// 128 threads (4 waves), 64-pixel tile per workgroup. grid = 8*256.
// ---------------------------------------------------------------------------
__global__ __launch_bounds__(128) void pp_out(const float* __restrict__ xg,
                                              const unsigned short* __restrict__ keysBg,
                                              const unsigned short* __restrict__ keysTg,
                                              float* __restrict__ outg) {
  __shared__ unsigned short Xt[64 * 264];     // x tile, row = pixel
  __shared__ unsigned short Kb[16 * 264];     // keys,  row = p (B for sim GEMM)
  __shared__ unsigned short Kt[256 * 24];     // keys^T row = c (B for out GEMM)
  __shared__ unsigned short sY[4 * 16 * 24];  // per-wave softmax(sim) tiles

  const int tid  = threadIdx.x;
  const int lane = tid & 31;
  const int wv   = tid >> 5;
  const int n    = blockIdx.x >> 8;           // 256 tiles per batch
  const int m0   = (blockIdx.x & 255) << 6;

  const float4* xrow = (const float4*)(xg + (size_t)(n * HW_DIM + m0) * C_DIM);
  unsigned int* XtU = (unsigned int*)Xt;
  for (int idx = tid; idx < 64 * 64; idx += 128) {
    int row = idx >> 6, c4 = idx & 63;
    float4 f = xrow[row * 64 + c4];
    int base = row * 264 + c4 * 4;
    XtU[(base >> 1) + 0] = pk2(f.x, f.y);
    XtU[(base >> 1) + 1] = pk2(f.z, f.w);
  }
  const unsigned int* kbU = (const unsigned int*)(keysBg + (n << 12));
  unsigned int* KbU = (unsigned int*)Kb;
  for (int idx = tid; idx < 2048; idx += 128) {   // 16 x 256 halves
    int row = idx >> 7, cp = idx & 127;
    KbU[row * 132 + cp] = kbU[idx];
  }
  const unsigned int* ktU = (const unsigned int*)(keysTg + (n << 12));
  unsigned int* KtU = (unsigned int*)Kt;
  for (int idx = tid; idx < 2048; idx += 128) {   // 256 x 16 halves
    int row = idx >> 3, cp = idx & 7;
    KtU[row * 12 + cp] = ktU[idx];
  }
  __syncthreads();

  // sim(16pix x 16p) = x @ keys^T, K = 256
  const int r0 = wv << 4;
  const int g = lane >> 4, m = lane & 15;
  v8f acc = {0, 0, 0, 0, 0, 0, 0, 0};
#pragma unroll
  for (int kc = 0; kc < 8; ++kc) {
    v16bf a = frag_ld(Xt, 264, r0, lane, kc * 32);
    v16bf b = frag_ld(Kb, 264, 0, lane, kc * 32);
    acc = wmma_bf16(a, b, acc);
  }

  // scaled softmax over p; write y into wave-private LDS tile [pix][p]
  unsigned short* sw = sY + wv * (16 * 24);
#pragma unroll
  for (int r = 0; r < 8; ++r) {
    float v = acc[r] * 0.0625f;                // 1/sqrt(256)
    float mx = red_max16(v);
    float e = __expf(v - mx);
    float s = red_sum16(e);
    sw[(r + (g << 3)) * 24 + m] = f2bf(e / s);
  }
  __syncthreads();

  // out(16pix x 256c) = y(16 x 16) @ keys(16 x 256); K=16 zero-padded to 32
  v16bf a_s = frag_ld_half(sw, 24, 0, lane);
  float* orow = outg + (size_t)(n * HW_DIM + m0 + r0) * C_DIM;
#pragma unroll
  for (int ct = 0; ct < 16; ++ct) {
    v16bf b = frag_ld_half(Kt, 24, ct << 4, lane);
    v8f z = {0, 0, 0, 0, 0, 0, 0, 0};
    v8f o = wmma_bf16(a_s, b, z);
#pragma unroll
    for (int r = 0; r < 8; ++r) {
      orow[(size_t)(r + (g << 3)) * C_DIM + (ct << 4) + m] = o[r];
    }
  }
}

// ---------------------------------------------------------------------------
extern "C" void kernel_launch(void* const* d_in, const int* in_sizes, int n_in,
                              void* d_out, int out_size, void* d_ws, size_t ws_size,
                              hipStream_t stream) {
  const float* xg = (const float*)d_in[0];   // (8,128,128,256) f32
  const float* wg = (const float*)d_in[1];   // (256,16) f32
  float* outg = (float*)d_out;               // (8,128,128,256) f32

  char* ws = (char*)d_ws;
  float* keysF          = (float*)ws;                           // 128 KB
  unsigned short* keysB = (unsigned short*)(ws + (131072));     //  64 KB
  unsigned short* keysT = (unsigned short*)(ws + (131072 + 65536));

  pp_zero<<<128, 256, 0, stream>>>(keysF);
  pp_keys<<<8 * 512, 64, 0, stream>>>(xg, wg, keysF);
  pp_cvt<<<128, 256, 0, stream>>>(keysF, keysB, keysT);
  pp_out<<<8 * 256, 128, 0, stream>>>(xg, keysB, keysT, outg);
}
